// Transformer_33543694582125
// MI455X (gfx1250) — matrix operand order repaired and checked by reading, not compiled, not run on hardware
//
#include <hip/hip_runtime.h>
#include <hip/hip_bf16.h>

typedef _Float16 h16;
typedef __attribute__((ext_vector_type(16))) _Float16 v16h;
typedef __attribute__((ext_vector_type(8)))  _Float16 v8h;
typedef __attribute__((ext_vector_type(8)))  float    v8f;

#define LAYERS 6
#define EDIM   768
#define SEQ    1024
#define VOCAB  32000
#define BATCH  2
#define TLEN   1024
#define FFDIM  3072
#define NTOK   (BATCH*TLEN)

#define BM 128
#define BN 128
#define BK 32
#define AST (BK+8)   /* padded LDS stride in halves: 80B rows, 16B aligned chunks */

static __device__ __forceinline__ unsigned pk2(float a, float b) {
    h16 ha = (h16)a, hb = (h16)b;
    unsigned short ua = __builtin_bit_cast(unsigned short, ha);
    unsigned short ub = __builtin_bit_cast(unsigned short, hb);
    return (unsigned)ua | ((unsigned)ub << 16);
}
static __device__ __forceinline__ unsigned short cvt1(float a) {
    h16 ha = (h16)a;
    return __builtin_bit_cast(unsigned short, ha);
}

// C = A(fp32, MxK) * B(fp32) + bias, via v_wmma_f32_16x16x32_f16.
// BT:     B stored [N][K] row-major (used for Q*K^T where B = K activations)
// RELU:   fused ReLU epilogue (FFN1)
// SKIPUP: causal scores — skip blocks entirely above the diagonal
// KCAP:   causal A*V — K-loop capped at m0+BM (probs beyond row are zero)
template<bool BT, bool RELU, bool SKIPUP, bool KCAP>
__global__ void __launch_bounds__(256)
gemm_f16_wmma(const float* __restrict__ A, const float* __restrict__ Bm,
              const float* __restrict__ bias, float* __restrict__ C,
              int M, int N, int K, int lda, int ldb, int ldc,
              long aBatch, long bBatch, long cBatch)
{
    __shared__ unsigned short sA[BM * AST];
    __shared__ unsigned short sB[BN * AST];

    const int m0 = blockIdx.y * BM;
    const int n0 = blockIdx.x * BN;
    if (SKIPUP && n0 > m0 + BM - 1) return;

    A  += (long)blockIdx.z * aBatch;
    Bm += (long)blockIdx.z * bBatch;
    C  += (long)blockIdx.z * cBatch;

    int kEnd = K;
    if (KCAP) { int ke = m0 + BM; kEnd = ke < K ? ke : K; }

    const int tid  = threadIdx.x;
    const int lane = tid & 31;
    const int wv   = tid >> 5;
    const int wm   = wv >> 1;            // 0..3 (M direction)
    const int wn   = wv & 1;             // 0..1 (N direction)
    const int lm   = lane & 15;
    const int khA  = (lane >> 4) * 8;    // A frag: chunks at khA and 16+khA (halves)
    const int khB  = (lane >> 4) * 16;   // unused: B frag uses same chunks as A (khA and 16+khA)

    const v8f vzero = {0.f, 0.f, 0.f, 0.f, 0.f, 0.f, 0.f, 0.f};
    v8f acc[2][4];
#pragma unroll
    for (int i = 0; i < 2; i++)
#pragma unroll
        for (int j = 0; j < 4; j++) acc[i][j] = vzero;

    const int la_c4  = (tid & 7) * 4;    // A/B(BT) tile loaders: 128 rows x 32 cols
    const int la_r   = tid >> 3;
    const int lb_cn  = (tid & 31) * 4;   // B(NN) tile loader: 32 k-rows x 128 n-cols
    const int lb_k   = tid >> 5;

    for (int k0 = 0; k0 < kEnd; k0 += BK) {
        if (k0 + BK < kEnd) {            // pull next slice toward L2 (global_prefetch_b8)
            __builtin_prefetch(A + (long)(m0 + la_r) * lda + (k0 + BK + la_c4), 0, 1);
            if (BT) __builtin_prefetch(Bm + (long)(n0 + la_r) * ldb + (k0 + BK + la_c4), 0, 1);
            else    __builtin_prefetch(Bm + (long)(k0 + BK + lb_k) * ldb + (n0 + lb_cn), 0, 1);
        }
        // ---- stage A tile (fp32 -> f16) ----
#pragma unroll
        for (int i = 0; i < 4; i++) {
            int row = la_r + i * 32;
            const float4 f = *(const float4*)(A + (long)(m0 + row) * lda + (k0 + la_c4));
            *(uint2*)&sA[row * AST + la_c4] = make_uint2(pk2(f.x, f.y), pk2(f.z, f.w));
        }
        // ---- stage B tile as [n][k] ----
        if (BT) {
#pragma unroll
            for (int i = 0; i < 4; i++) {
                int row = la_r + i * 32;
                const float4 f = *(const float4*)(Bm + (long)(n0 + row) * ldb + (k0 + la_c4));
                *(uint2*)&sB[row * AST + la_c4] = make_uint2(pk2(f.x, f.y), pk2(f.z, f.w));
            }
        } else {
#pragma unroll
            for (int i = 0; i < 4; i++) {
                int kk = lb_k + i * 8;
                const float4 f = *(const float4*)(Bm + (long)(k0 + kk) * ldb + (n0 + lb_cn));
                sB[(lb_cn + 0) * AST + kk] = cvt1(f.x);
                sB[(lb_cn + 1) * AST + kk] = cvt1(f.y);
                sB[(lb_cn + 2) * AST + kk] = cvt1(f.z);
                sB[(lb_cn + 3) * AST + kk] = cvt1(f.w);
            }
        }
        __syncthreads();

        // ---- fragments per ISA layout ----
        v16h af[2], bf[4];
#pragma unroll
        for (int mt = 0; mt < 2; mt++) {
            int r = wm * 32 + mt * 16 + lm;
            v8h lo = *(const v8h*)&sA[r * AST + khA];
            v8h hi = *(const v8h*)&sA[r * AST + 16 + khA];
            af[mt] = __builtin_shufflevector(lo, hi, 0,1,2,3,4,5,6,7,8,9,10,11,12,13,14,15);
        }
#pragma unroll
        for (int nt = 0; nt < 4; nt++) {
            int r = wn * 64 + nt * 16 + lm;
            v8h lo = *(const v8h*)&sB[r * AST + khA];
            v8h hi = *(const v8h*)&sB[r * AST + 16 + khA];
            bf[nt] = __builtin_shufflevector(lo, hi, 0,1,2,3,4,5,6,7,8,9,10,11,12,13,14,15);
        }
#pragma unroll
        for (int mt = 0; mt < 2; mt++)
#pragma unroll
            for (int nt = 0; nt < 4; nt++)
                acc[mt][nt] = __builtin_amdgcn_wmma_f32_16x16x32_f16(
                    false, af[mt], false, bf[nt], (short)0, acc[mt][nt], false, false);
        __syncthreads();
    }

    // ---- epilogue: C/D layout row = i + 8*(lane>>4), col = lane&15 ----
    const int colBase = n0 + wn * 64 + lm;
    const int rowBase = m0 + wm * 32 + (lane >> 4) * 8;
#pragma unroll
    for (int nt = 0; nt < 4; nt++) {
        int col = colBase + nt * 16;
        float bvv = bias ? bias[col] : 0.f;
#pragma unroll
        for (int mt = 0; mt < 2; mt++) {
            int r0 = rowBase + mt * 16;
#pragma unroll
            for (int i = 0; i < 8; i++) {
                float vv = acc[mt][nt][i] + bvv;
                if (RELU) vv = fmaxf(vv, 0.f);
                C[(long)(r0 + i) * ldc + col] = vv;
            }
        }
    }
}

// causal softmax in-place over scores[B*T][T]; the reference's "+1.0 on allowed
// positions" is a constant shift inside softmax and cancels exactly.
__global__ void __launch_bounds__(256)
softmax_causal(float* __restrict__ sc)
{
    const int row = blockIdx.x;
    const int t   = row & (TLEN - 1);
    float* p = sc + (long)row * TLEN;
    const int n = t + 1;
    const float is = 0.03608439182435161f; // 1/sqrt(768)
    const int tid = threadIdx.x, lane = tid & 31, wid = tid >> 5;
    __shared__ float red[8];

    float mx = -3.4e38f;
    for (int s = tid; s < n; s += 256) mx = fmaxf(mx, p[s] * is);
#pragma unroll
    for (int off = 16; off > 0; off >>= 1) mx = fmaxf(mx, __shfl_xor(mx, off, 32));
    if (lane == 0) red[wid] = mx;
    __syncthreads();
    if (tid == 0) { float r = red[0]; for (int i = 1; i < 8; i++) r = fmaxf(r, red[i]); red[0] = r; }
    __syncthreads();
    mx = red[0];
    __syncthreads();

    float sum = 0.f;
    for (int s = tid; s < n; s += 256) { float e = __expf(p[s] * is - mx); p[s] = e; sum += e; }
#pragma unroll
    for (int off = 16; off > 0; off >>= 1) sum += __shfl_xor(sum, off, 32);
    if (lane == 0) red[wid] = sum;
    __syncthreads();
    if (tid == 0) { float r = 0.f; for (int i = 0; i < 8; i++) r += red[i]; red[0] = r; }
    __syncthreads();
    const float rinv = 1.f / red[0];
    for (int s = tid; s < n; s += 256) p[s] *= rinv;
    for (int s = n + tid - (tid < ((n) & 255) ? 0 : 0); s < TLEN; s += 256) if (s >= n) p[s] = 0.f;
}

// out = LayerNorm(a + b) * w + bias   (E=768 = 256*3)
__global__ void __launch_bounds__(256)
add_layernorm(const float* __restrict__ a, const float* __restrict__ b,
              const float* __restrict__ w, const float* __restrict__ bb,
              float* __restrict__ out)
{
    const int row = blockIdx.x;
    const int tid = threadIdx.x, lane = tid & 31, wid = tid >> 5;
    __shared__ float r1[8], r2[8];
    float x[3]; float s = 0.f, s2 = 0.f;
#pragma unroll
    for (int j = 0; j < 3; j++) {
        int c = tid + j * 256;
        float v = a[(long)row * EDIM + c] + b[(long)row * EDIM + c];
        x[j] = v; s += v; s2 += v * v;
    }
#pragma unroll
    for (int off = 16; off > 0; off >>= 1) { s += __shfl_xor(s, off, 32); s2 += __shfl_xor(s2, off, 32); }
    if (lane == 0) { r1[wid] = s; r2[wid] = s2; }
    __syncthreads();
    if (tid == 0) {
        float a1 = 0.f, a2 = 0.f;
        for (int i = 0; i < 8; i++) { a1 += r1[i]; a2 += r2[i]; }
        r1[0] = a1; r2[0] = a2;
    }
    __syncthreads();
    const float mean = r1[0] * (1.f / EDIM);
    const float var  = r2[0] * (1.f / EDIM) - mean * mean;
    const float rstd = rsqrtf(var + 1e-5f);
#pragma unroll
    for (int j = 0; j < 3; j++) {
        int c = tid + j * 256;
        out[(long)row * EDIM + c] = (x[j] - mean) * rstd * w[c] + bb[c];
    }
}

__global__ void __launch_bounds__(256)
embed_kernel(const int* __restrict__ x, const float* __restrict__ tok,
             const float* __restrict__ pos, float* __restrict__ h)
{
    const int row = blockIdx.x;
    const int t = row & (TLEN - 1);
    const int tk = x[row];
#pragma unroll
    for (int j = 0; j < 3; j++) {
        int c = threadIdx.x + j * 256;
        h[(long)row * EDIM + c] = tok[(long)tk * EDIM + c] + pos[(long)t * EDIM + c];
    }
}

extern "C" void kernel_launch(void* const* d_in, const int* in_sizes, int n_in,
                              void* d_out, int out_size, void* d_ws, size_t ws_size,
                              hipStream_t stream)
{
    const int*   x    = (const int*)d_in[0];
    const float* tok  = (const float*)d_in[1];
    const float* pos  = (const float*)d_in[2];
    const float* Wq   = (const float*)d_in[3];  const float* bq   = (const float*)d_in[4];
    const float* Wk   = (const float*)d_in[5];  const float* bk   = (const float*)d_in[6];
    const float* Wv   = (const float*)d_in[7];  const float* bv   = (const float*)d_in[8];
    const float* ln1w = (const float*)d_in[9];  const float* ln1b = (const float*)d_in[10];
    const float* W1   = (const float*)d_in[11]; const float* b1   = (const float*)d_in[12];
    const float* W2   = (const float*)d_in[13]; const float* b2   = (const float*)d_in[14];
    const float* ln2w = (const float*)d_in[15]; const float* ln2b = (const float*)d_in[16];
    const float* Wo   = (const float*)d_in[17]; const float* bo   = (const float*)d_in[18];
    float* out = (float*)d_out;

    float* ws = (float*)d_ws;
    const long HS = (long)NTOK * EDIM;
    float* h   = ws; ws += HS;
    float* q   = ws; ws += HS;
    float* k   = ws; ws += HS;
    float* v   = ws; ws += HS;
    float* att = ws; ws += HS;            // also reused for FFN2 output
    float* h1  = ws; ws += HS;
    float* sc  = ws; ws += (long)BATCH * TLEN * TLEN;
    float* f1  = ws; ws += (long)NTOK * FFDIM;
    (void)ws_size; (void)n_in; (void)in_sizes; (void)out_size;

    embed_kernel<<<NTOK, 256, 0, stream>>>(x, tok, pos, h);

    const dim3 blk(256);
    for (int l = 0; l < LAYERS; l++) {
        const float* wq = Wq + (long)l * EDIM * EDIM;
        const float* wk = Wk + (long)l * EDIM * EDIM;
        const float* wv = Wv + (long)l * EDIM * EDIM;

        dim3 gEE(EDIM / BN, NTOK / BM, 1);
        gemm_f16_wmma<false,false,false,false><<<gEE, blk, 0, stream>>>(
            h, wq, bq + (long)l * EDIM, q, NTOK, EDIM, EDIM, EDIM, EDIM, EDIM, 0, 0, 0);
        gemm_f16_wmma<false,false,false,false><<<gEE, blk, 0, stream>>>(
            h, wk, bk + (long)l * EDIM, k, NTOK, EDIM, EDIM, EDIM, EDIM, EDIM, 0, 0, 0);
        gemm_f16_wmma<false,false,false,false><<<gEE, blk, 0, stream>>>(
            h, wv, bv + (long)l * EDIM, v, NTOK, EDIM, EDIM, EDIM, EDIM, EDIM, 0, 0, 0);

        // scores = Q * K^T  (per batch; upper-triangle blocks skipped)
        dim3 gSc(TLEN / BN, TLEN / BM, BATCH);
        gemm_f16_wmma<true,false,true,false><<<gSc, blk, 0, stream>>>(
            q, k, nullptr, sc, TLEN, TLEN, EDIM, EDIM, EDIM, TLEN,
            (long)TLEN * EDIM, (long)TLEN * EDIM, (long)TLEN * TLEN);

        softmax_causal<<<NTOK, 256, 0, stream>>>(sc);

        // att = P * V  (K-loop capped causally)
        dim3 gAV(EDIM / BN, TLEN / BM, BATCH);
        gemm_f16_wmma<false,false,false,true><<<gAV, blk, 0, stream>>>(
            sc, v, nullptr, att, TLEN, EDIM, TLEN, TLEN, EDIM, EDIM,
            (long)TLEN * TLEN, (long)TLEN * EDIM, (long)TLEN * EDIM);

        add_layernorm<<<NTOK, 256, 0, stream>>>(h, att, ln1w + (long)l * EDIM,
                                                ln1b + (long)l * EDIM, h1);

        dim3 gF1(FFDIM / BN, NTOK / BM, 1);
        gemm_f16_wmma<false,true,false,false><<<gF1, blk, 0, stream>>>(
            h1, W1 + (long)l * EDIM * FFDIM, b1 + (long)l * FFDIM, f1,
            NTOK, FFDIM, EDIM, EDIM, FFDIM, FFDIM, 0, 0, 0);

        dim3 gF2(EDIM / BN, NTOK / BM, 1);
        gemm_f16_wmma<false,false,false,false><<<gF2, blk, 0, stream>>>(
            f1, W2 + (long)l * FFDIM * EDIM, b2 + (long)l * EDIM, att,
            NTOK, EDIM, FFDIM, FFDIM, EDIM, EDIM, 0, 0, 0);

        add_layernorm<<<NTOK, 256, 0, stream>>>(att, h1, ln2w + (long)l * EDIM,
                                                ln2b + (long)l * EDIM, h);
    }

    // logits = h @ Wo + bo   (N = 32000 = 250*128)
    dim3 gO(VOCAB / BN, NTOK / BM, 1);
    gemm_f16_wmma<false,false,false,false><<<gO, blk, 0, stream>>>(
        h, Wo, bo, out, NTOK, VOCAB, EDIM, EDIM, VOCAB, VOCAB, 0, 0, 0);
}
